// MonitoredMultiHeadAttention_59030030516986
// MI455X (gfx1250) — matrix operand order repaired and checked by reading, not compiled, not run on hardware
//
#include <hip/hip_runtime.h>
#include <hip/hip_bf16.h>

// MI455X (gfx1250) fused multi-head attention.
// f16 operands + f32 accumulation via V_WMMA_F32_16X16X32_F16, async global->LDS
// staging (GLOBAL_LOAD_ASYNC_TO_LDS_B128 / ASYNCcnt) with double-buffered K panels.
// Pipeline:
//   [cvt f32->f16: x, Wq..Wo] -> [gemm x3 -> Qh,Kh,Vh (f16)]
//   -> [flash attention per (b,h) -> AOh (f16)] -> [gemm Wo -> d_out (f32)]
// Workspace (f16): xh | wqh | wkh | wvh | woh | Qh | Kh | Vh | AOh  (~48 MB).

#define D_MODEL 1024
#define SEQ     2048
#define BATCH   2
#define NHEADS  16
#define DK      64

typedef _Float16 v4h  __attribute__((ext_vector_type(4)));
typedef _Float16 v8h  __attribute__((ext_vector_type(8)));
typedef _Float16 v16h __attribute__((ext_vector_type(16)));
typedef float    v8f  __attribute__((ext_vector_type(8)));
typedef int      v4i_ __attribute__((vector_size(16)));   // matches builtin param pointee

// ---- async global->LDS (CDNA5), guarded so absence degrades to sync copies ----
#if defined(__has_builtin)
# if __has_builtin(__builtin_amdgcn_global_load_async_to_lds_b128) && \
     __has_builtin(__builtin_amdgcn_s_wait_asynccnt)
#  define HAVE_ASYNC_LDS 1
# endif
#endif
#ifndef HAVE_ASYNC_LDS
# define HAVE_ASYNC_LDS 0
#endif

__device__ __forceinline__ void cp_b128(void* lds, const void* g) {
#if HAVE_ASYNC_LDS
    // AS1 (global): flat address == global address.
    // AS3 (LDS): generic LDS pointer carries the LDS offset in its low 32 bits.
    __builtin_amdgcn_global_load_async_to_lds_b128(
        (__attribute__((address_space(1))) v4i_*)(unsigned long long)g,
        (__attribute__((address_space(3))) v4i_*)(unsigned int)(unsigned long long)lds,
        0, 0);
#else
    *(uint4*)lds = *(const uint4*)g;
#endif
}
template<int N>
__device__ __forceinline__ void cp_wait() {
#if HAVE_ASYNC_LDS
    __builtin_amdgcn_s_wait_asynccnt(N);
#endif
}

__device__ __forceinline__ v8f wmma_f16(v16h a, v16h b, v8f c) {
    return __builtin_amdgcn_wmma_f32_16x16x32_f16(
        false, a, false, b, (short)0, c, false, false);
}

__device__ __forceinline__ v16h cat16(v8h lo, v8h hi) {
    return __builtin_shufflevector(lo, hi, 0,1,2,3,4,5,6,7,8,9,10,11,12,13,14,15);
}

// A-fragment (16x32 f16, ISA 7.12.2): lane(r,half) holds K = 8*half+[0..7] and
// K = 16+8*half+[0..7] of row r.
__device__ __forceinline__ v16h afrag(const _Float16* p, int kbase, int half) {
    v8h lo = *(const v8h*)(p + kbase + half * 8);
    v8h hi = *(const v8h*)(p + kbase + 16 + half * 8);
    return cat16(lo, hi);
}
// B-fragment (32x16 f16): lane(r,half) holds K = 8*half+[0..7] and K = 16+8*half+[0..7] of col r.
// With B[k][n] = src[n][k] (row-major [N][K] source), these are two 16-byte runs.
__device__ __forceinline__ v16h bfrag(const _Float16* p, int kbase, int half) {
    v8h lo = *(const v8h*)(p + kbase + half * 8);
    v8h hi = *(const v8h*)(p + kbase + 16 + half * 8);
    return cat16(lo, hi);
}

// ---------------------------------------------------------------------------
// f32 -> f16 conversion (elementwise, vectorized)
// ---------------------------------------------------------------------------
__global__ __launch_bounds__(256) void cvt_f32_f16_kernel(
    const float* __restrict__ src, _Float16* __restrict__ dst, int n)
{
    int i = (blockIdx.x * 256 + threadIdx.x) * 4;
    if (i < n) {
        float4 v = *(const float4*)(src + i);
        v4h o = { (_Float16)v.x, (_Float16)v.y, (_Float16)v.z, (_Float16)v.w };
        *(v4h*)(dst + i) = o;
    }
}

// ---------------------------------------------------------------------------
// GEMM: C[M x N] = A[M x K] * W[N x K]^T, f16 operands, f32 accumulate.
// Block tile 128x64, 8 waves (4x2), wave tile 32x32 (4 accumulators), KB=32.
// Double-buffered LDS panels with async global->LDS staging.
// ---------------------------------------------------------------------------
#define GBM 128
#define GBN 64
#define GKB 32
#define ALD 40   // padded LDS row stride (halfwords): 80B rows, conflict-free, 16B aligned

template<bool F16OUT>
__global__ __launch_bounds__(256) void gemm_f16_kernel(
    const _Float16* __restrict__ A, const _Float16* __restrict__ W,
    void* __restrict__ Cv, int M, int N, int K)
{
    __shared__ _Float16 As[2][GBM * ALD];
    __shared__ _Float16 Ws[2][GBN * ALD];

    const int tid  = threadIdx.x;
    const int lane = tid & 31;
    const int wave = tid >> 5;
    const int half = lane >> 4;
    const int r    = lane & 15;
    const int wr   = wave >> 1;       // 0..3 : 32-row slab
    const int wc   = wave & 1;        // 0..1 : 32-col slab
    const int m0   = blockIdx.y * GBM;
    const int n0   = blockIdx.x * GBN;

    // Per-thread staging coordinates (3 x b128 per thread per panel).
    const int ar0 = (tid)       >> 2, ac0 = ((tid)       & 3) << 3;
    const int ar1 = (tid + 256) >> 2, ac1 = ((tid + 256) & 3) << 3;
    const int wrw = tid >> 2,         wcw = (tid & 3) << 3;

    auto stage = [&](int buf, int k0) {
        cp_b128(&As[buf][ar0 * ALD + ac0], A + (size_t)(m0 + ar0) * K + k0 + ac0);
        cp_b128(&As[buf][ar1 * ALD + ac1], A + (size_t)(m0 + ar1) * K + k0 + ac1);
        cp_b128(&Ws[buf][wrw * ALD + wcw], W + (size_t)(n0 + wrw) * K + k0 + wcw);
    };

    v8f acc[2][2] = {};
    stage(0, 0);
    const int nIter = K / GKB;

    for (int i = 0; i < nIter; ++i) {
        __syncthreads();                       // all reads of buf[i^1] retired
        const bool more = (i + 1) < nIter;
        if (more) { stage((i + 1) & 1, (i + 1) * GKB); cp_wait<3>(); }
        else      { cp_wait<0>(); }
        __syncthreads();

        const _Float16* as = As[i & 1];
        const _Float16* ws = Ws[i & 1];
        v16h a0 = afrag(&as[(wr * 32 +      r) * ALD], 0, half);
        v16h a1 = afrag(&as[(wr * 32 + 16 + r) * ALD], 0, half);
        v16h b0 = bfrag(&ws[(wc * 32 +      r) * ALD], 0, half);
        v16h b1 = bfrag(&ws[(wc * 32 + 16 + r) * ALD], 0, half);
        acc[0][0] = wmma_f16(a0, b0, acc[0][0]);
        acc[0][1] = wmma_f16(a0, b1, acc[0][1]);
        acc[1][0] = wmma_f16(a1, b0, acc[1][0]);
        acc[1][1] = wmma_f16(a1, b1, acc[1][1]);
    }

    // C/D layout: VGPR v holds M = v + 8*half, N = lane%16
#pragma unroll
    for (int i = 0; i < 2; ++i)
#pragma unroll
    for (int j = 0; j < 2; ++j)
#pragma unroll
    for (int v = 0; v < 8; ++v) {
        size_t row = m0 + wr * 32 + i * 16 + v + half * 8;
        size_t col = n0 + wc * 32 + j * 16 + r;
        if (F16OUT) ((_Float16*)Cv)[row * N + col] = (_Float16)acc[i][j][v];
        else        ((float*)Cv)[row * N + col]    = acc[i][j][v];
    }
}

// ---------------------------------------------------------------------------
// Flash attention per (b,h): block owns 128 query rows (8 waves x 16 rows),
// streams keys/values in 64-wide tiles, online softmax, f16 WMMA for QK^T / PV.
// K tile staged via async LDS copies while the V transpose runs on the VALU.
// ---------------------------------------------------------------------------
#define BQ   128
#define BKV  64
#define KLD  72   // Ks [64 kv][64 dk + pad]  : 144B rows, conflict-free, 16B aligned
#define VTLD 72   // Vt [64 dk][64 kv + pad]
#define PLD  72   // Ps [128 q ][64 kv + pad]

__global__ __launch_bounds__(256) void attn_kernel(
    const _Float16* __restrict__ Q, const _Float16* __restrict__ Km,
    const _Float16* __restrict__ V, _Float16* __restrict__ O)
{
    __shared__ _Float16 Ks[BKV * KLD];
    __shared__ _Float16 Vt[DK  * VTLD];
    __shared__ _Float16 Ps[BQ  * PLD];

    const int tid  = threadIdx.x;
    const int lane = tid & 31;
    const int wave = tid >> 5;        // 0..7 -> 16 query rows each
    const int half = lane >> 4;
    const int r    = lane & 15;
    const int bh   = blockIdx.y;      // b*16 + h
    const int b    = bh >> 4;
    const int h    = bh & 15;
    const int q0   = blockIdx.x * BQ;
    const size_t rowBase = (size_t)b * SEQ;
    const int hc   = h * DK;

    // Preload this wave's 16x64 Q block as ready-made A fragments (registers).
    v16h qf[2];
    {
        const _Float16* qp = Q + (rowBase + q0 + wave * 16 + r) * D_MODEL + hc;
        qf[0] = afrag(qp, 0,  half);
        qf[1] = afrag(qp, 32, half);
    }

    v8f oacc[4] = {};       // 16 rows x 64 cols output accumulator (f32)
    float rmax[8], rsum[8]; // online-softmax stats: row = v + 8*half
#pragma unroll
    for (int v = 0; v < 8; ++v) { rmax[v] = -3.0e38f; rsum[v] = 0.0f; }

    for (int kv0 = 0; kv0 < SEQ; kv0 += BKV) {
        __syncthreads();   // protect Ks/Vt/Ps from previous iteration's readers
        // Async-stage K tile; overlap with manual V transpose (Vt[dk][kv]).
#pragma unroll
        for (int it = 0; it < 2; ++it) {
            int slot = tid + it * 256;   // 512 chunks of 8 f16
            int row  = slot >> 3;        // kv index 0..63
            int c8   = (slot & 7) << 3;  // dk offset 0..56
            const size_t g = (rowBase + kv0 + row) * D_MODEL + hc + c8;
            cp_b128(&Ks[row * KLD + c8], Km + g);
            uint4 vv = *(const uint4*)(V + g);
            v8h vh = *(const v8h*)&vv;
#pragma unroll
            for (int e = 0; e < 8; ++e)   // scatter-transpose into Vt
                Vt[(c8 + e) * VTLD + row] = vh[e];
        }
        cp_wait<0>();
        __syncthreads();

        // S(16x64) = Qw(16x64) * K^T(64x64): B[k][n] = K[n][k] -> rows of Ks.
        v8f sacc[4] = {};
#pragma unroll
        for (int kk = 0; kk < 2; ++kk) {
#pragma unroll
            for (int j = 0; j < 4; ++j) {
                v16h bv = bfrag(&Ks[(j * 16 + r) * KLD], kk * 32, half);
                sacc[j] = wmma_f16(qf[kk], bv, sacc[j]);
            }
        }

        // Online softmax over this 64-key slab; row stats via 16-lane xor reduce.
#pragma unroll
        for (int v = 0; v < 8; ++v) {
            float s[4];
#pragma unroll
            for (int j = 0; j < 4; ++j) s[j] = sacc[j][v] * 0.125f;  // 1/sqrt(Dk)
            float m = fmaxf(fmaxf(s[0], s[1]), fmaxf(s[2], s[3]));
#pragma unroll
            for (int off = 1; off < 16; off <<= 1)
                m = fmaxf(m, __shfl_xor(m, off, 32));
            float nm    = fmaxf(rmax[v], m);
            float scale = __expf(rmax[v] - nm);
            float p[4], ps = 0.0f;
#pragma unroll
            for (int j = 0; j < 4; ++j) { p[j] = __expf(s[j] - nm); ps += p[j]; }
#pragma unroll
            for (int off = 1; off < 16; off <<= 1)
                ps += __shfl_xor(ps, off, 32);
            rsum[v] = rsum[v] * scale + ps;
            rmax[v] = nm;
#pragma unroll
            for (int j = 0; j < 4; ++j) oacc[j][v] *= scale;
            // P (f16) into LDS, A-fragment friendly row-major layout.
            int prow = wave * 16 + v + half * 8;
#pragma unroll
            for (int j = 0; j < 4; ++j)
                Ps[prow * PLD + j * 16 + r] = (_Float16)p[j];
        }
        __syncthreads();

        // O(16x64) += P(16x64) * V(64x64); B[k][n] = V[k][n] -> rows of Vt.
#pragma unroll
        for (int kk = 0; kk < 2; ++kk) {
            v16h a = afrag(&Ps[(wave * 16 + r) * PLD], kk * 32, half);
#pragma unroll
            for (int j = 0; j < 4; ++j) {
                v16h bv = bfrag(&Vt[(j * 16 + r) * VTLD], kk * 32, half);
                oacc[j] = wmma_f16(a, bv, oacc[j]);
            }
        }
    }

    // Normalize and write attention output (f16, heads concatenated).
#pragma unroll
    for (int v = 0; v < 8; ++v) {
        float inv = 1.0f / rsum[v];
        size_t row = q0 + wave * 16 + v + half * 8;
        _Float16* op = O + (rowBase + row) * D_MODEL + hc;
#pragma unroll
        for (int j = 0; j < 4; ++j)
            op[j * 16 + r] = (_Float16)(oacc[j][v] * inv);
    }
}

// ---------------------------------------------------------------------------
extern "C" void kernel_launch(void* const* d_in, const int* in_sizes, int n_in,
                              void* d_out, int out_size, void* d_ws, size_t ws_size,
                              hipStream_t stream) {
    const float* x  = (const float*)d_in[0];
    const float* Wq = (const float*)d_in[1];
    const float* Wk = (const float*)d_in[2];
    const float* Wv = (const float*)d_in[3];
    const float* Wo = (const float*)d_in[4];

    const int M = BATCH * SEQ;                 // 4096
    const size_t SZ = (size_t)M * D_MODEL;     // 4,194,304
    const size_t WZ = (size_t)D_MODEL * D_MODEL;

    _Float16* xh  = (_Float16*)d_ws;
    _Float16* wqh = xh  + SZ;
    _Float16* wkh = wqh + WZ;
    _Float16* wvh = wkh + WZ;
    _Float16* woh = wvh + WZ;
    _Float16* Qh  = woh + WZ;
    _Float16* Kh  = Qh  + SZ;
    _Float16* Vh  = Kh  + SZ;
    _Float16* AOh = Vh  + SZ;                  // total ~48 MB of d_ws

    cvt_f32_f16_kernel<<<(int)(SZ / 4 / 256), 256, 0, stream>>>(x,  xh,  (int)SZ);
    cvt_f32_f16_kernel<<<(int)(WZ / 4 / 256), 256, 0, stream>>>(Wq, wqh, (int)WZ);
    cvt_f32_f16_kernel<<<(int)(WZ / 4 / 256), 256, 0, stream>>>(Wk, wkh, (int)WZ);
    cvt_f32_f16_kernel<<<(int)(WZ / 4 / 256), 256, 0, stream>>>(Wv, wvh, (int)WZ);
    cvt_f32_f16_kernel<<<(int)(WZ / 4 / 256), 256, 0, stream>>>(Wo, woh, (int)WZ);

    dim3 gGrid(D_MODEL / GBN, M / GBM);        // (16, 32)
    gemm_f16_kernel<true><<<gGrid, 256, 0, stream>>>(xh, wqh, Qh, M, D_MODEL, D_MODEL);
    gemm_f16_kernel<true><<<gGrid, 256, 0, stream>>>(xh, wkh, Kh, M, D_MODEL, D_MODEL);
    gemm_f16_kernel<true><<<gGrid, 256, 0, stream>>>(xh, wvh, Vh, M, D_MODEL, D_MODEL);

    dim3 aGrid(SEQ / BQ, BATCH * NHEADS);      // (16, 32)
    attn_kernel<<<aGrid, 256, 0, stream>>>(Qh, Kh, Vh, AOh);

    gemm_f16_kernel<false><<<gGrid, 256, 0, stream>>>(AOh, woh, d_out, M, D_MODEL, D_MODEL);
}
